// BeatsRandomTokenizer_1614907703805
// MI455X (gfx1250) — matrix operand order repaired and checked by reading, not compiled, not run on hardware
//
#include <hip/hip_runtime.h>
#include <hip/hip_bf16.h>

// ---------------------------------------------------------------------------
// BEATs random tokenizer, fully fused for MI455X (gfx1250, wave32, WMMA).
//   fbank(16,8192,128) -> patchify conv(512) -> LN -> proj(256) -> l2norm
//   -> cosine vs l2norm codebook(1024) -> argmax  (65536 int indices)
// One WG = 64 tokens; all intermediates live in LDS (~265KB of the 320KB/WGP).
// GEMMs: v_wmma_f32_16x16x32_bf16 (fp32 accumulate), one A-frag shared
// across 8 N-tiles per wave. Weight streaming uses
// global_load_async_to_lds_b128 (ASYNCcnt) from a bf16-packed workspace.
// ---------------------------------------------------------------------------

#define THREADS   256           // 8 wave32s
#define TOK_PER_WG 64
#define NTOK      65536         // 16 * 4096 tokens

typedef __attribute__((ext_vector_type(16))) __bf16 v16bf;
typedef __attribute__((ext_vector_type(8)))  float  v8f;

union FragBF { v16bf v; uint4 q[2]; };

// ---- LDS layout (bytes). Row strides padded (+8 bf16) => odd*4-dword rows,
//      so 16-lane fragment loads hit distinct banks. All rows 16B aligned.
#define SA1 264                 // patch tile  [64][256] (+pad)
#define SFE 520                 // feats bf16  [64][512] (+pad)
#define SPB 264                 // p bf16      [64][256] (+pad)
#define SW1 264                 // conv-w chunk [256][256]
#define SW2 520                 // proj chunk   [128][512]
#define SW3 264                 // codebook chunk [256][256]
#define OFF_A1   0
#define OFF_FE   33792          // 64*264*2
#define OFF_PB   100352         // +64*520*2
#define OFF_WB   134144         // +64*264*2
#define OFF_STAT 269312         // +256*264*2 (max weight chunk)
#define LDS_TOTAL 271360        // +2KB stats

// workspace layout (bf16, [N][K] = WMMA-B friendly)
#define WS_CBN 0                // [1024][256] normalized codebook
#define WS_WB  (1024*256)       // [512][256]  conv weight (natural layout)
#define WS_PB  (1024*256 + 512*256)  // [256][512] proj transposed
#define WS_NEED ((size_t)(1024*256 + 512*256 + 256*512) * 2)

__device__ __forceinline__ unsigned short f2bf(float f) {
    unsigned u = __float_as_uint(f);
    u += 0x7FFFu + ((u >> 16) & 1u);          // round-to-nearest-even
    return (unsigned short)(u >> 16);
}
__device__ __forceinline__ float bf2f(unsigned short h) {
    return __uint_as_float(((unsigned)h) << 16);
}
// monotonic (totally ordered) unsigned key for a float
__device__ __forceinline__ unsigned fkey(float f) {
    unsigned u = __float_as_uint(f);
    return ((int)u < 0) ? ~u : (u | 0x80000000u);
}
__device__ __forceinline__ unsigned long long shfl_xor_u64_16(unsigned long long x, int m) {
    unsigned lo = __shfl_xor((unsigned)x, m, 16);
    unsigned hi = __shfl_xor((unsigned)(x >> 32), m, 16);
    return ((unsigned long long)hi << 32) | lo;
}

// per-lane async 16B global -> LDS copy (GLOBAL_LOAD_ASYNC_TO_LDS_B128, GV mode)
__device__ __forceinline__ void async_cp16(unsigned lds_byte_addr, const void* gsrc) {
    asm volatile("global_load_async_to_lds_b128 %0, %1, off"
                 :: "v"(lds_byte_addr), "v"((unsigned long long)(uintptr_t)gsrc)
                 : "memory");
}
__device__ __forceinline__ void wait_async0() {
    asm volatile("s_wait_asynccnt 0x0" ::: "memory");
}

// A fragment: 16x32 bf16, rows = M.  Lane l: row M=l&15; K-half (l>>4)*8,
// VGPRs 0..3 = K[kh..kh+7], VGPRs 4..7 = K[kh+16..kh+23]  (ISA 7.12.2).
__device__ __forceinline__ v16bf load_fragA(const unsigned short* base, int stride,
                                            int row0, int kbase, int lane) {
    const unsigned short* p = base + (row0 + (lane & 15)) * stride
                                   + kbase + ((lane >> 4) * 8);
    FragBF f;
    f.q[0] = *(const uint4*)(p);
    f.q[1] = *(const uint4*)(p + 16);
    return f.v;
}
// B fragment: 32x16 bf16 staged as [N][K]. Lane l: col N=l&15; K-half (l>>4)*8,
// VGPRs 0..3 = K[kh..kh+7], VGPRs 4..7 = K[kh+16..kh+23] (matches A).
__device__ __forceinline__ v16bf load_fragB(const unsigned short* base, int stride,
                                            int n0, int kbase, int lane) {
    const unsigned short* p = base + (n0 + (lane & 15)) * stride
                                   + kbase + ((lane >> 4) * 8);
    FragBF f;
    f.q[0] = *(const uint4*)(p);
    f.q[1] = *(const uint4*)(p + 16);
    return f.v;
}

// --- prep: l2-normalize codebook rows -> bf16 in workspace (one wave / row)
__global__ __launch_bounds__(256)
void normalize_codebook(const float* __restrict__ codebook,
                        unsigned short* __restrict__ cbn) {
    int wv = threadIdx.x >> 5, lane = threadIdx.x & 31;
    int row = blockIdx.x * 8 + wv;
    const float* src = codebook + (size_t)row * 256 + lane * 8;
    float x[8]; float s = 0.f;
#pragma unroll
    for (int i = 0; i < 8; ++i) { x[i] = src[i]; s += x[i] * x[i]; }
#pragma unroll
    for (int m = 1; m < 32; m <<= 1) s += __shfl_xor(s, m, 32);
    float sc = 1.f / fmaxf(sqrtf(s), 1e-12f);
    unsigned short* dst = cbn + (size_t)row * 256 + lane * 8;
#pragma unroll
    for (int i = 0; i < 8; ++i) dst[i] = f2bf(x[i] * sc);
}

// --- prep: bf16-pack conv weight (copy) and proj (transpose to [q][d])
__global__ __launch_bounds__(256)
void pack_weights(const float* __restrict__ conv_w, const float* __restrict__ proj,
                  unsigned short* __restrict__ wb, unsigned short* __restrict__ pb) {
    int e = blockIdx.x * 256 + threadIdx.x;          // 0 .. 131071
    wb[e] = f2bf(conv_w[e]);                         // [d][k] natural
    int d = e >> 8, q = e & 255;                     // coalesced read of proj
    pb[q * 512 + d] = f2bf(proj[e]);                 // [q][d] transposed
}

template <bool USE_WS>
__global__ __launch_bounds__(THREADS)
void beats_tokenizer_fused(const float* __restrict__ fbank,
                           const float* __restrict__ conv_w,
                           const float* __restrict__ proj,
                           const float* __restrict__ codebook,
                           const unsigned short* __restrict__ wsb,  // bf16 ws base
                           int* __restrict__ out) {
    extern __shared__ unsigned char smem[];
    unsigned short* A1 = (unsigned short*)(smem + OFF_A1);
    unsigned short* FE = (unsigned short*)(smem + OFF_FE);
    unsigned short* PB = (unsigned short*)(smem + OFF_PB);
    unsigned short* WB = (unsigned short*)(smem + OFF_WB);
    float* lnSum = (float*)(smem + OFF_STAT);
    float* lnSq  = (float*)(smem + OFF_STAT + 256);
    float* muA   = (float*)(smem + OFF_STAT + 512);
    float* rsA   = (float*)(smem + OFF_STAT + 768);
    float* psq   = (float*)(smem + OFF_STAT + 1024);
    float* prs   = (float*)(smem + OFF_STAT + 1280);
    unsigned long long* best = (unsigned long long*)(smem + OFF_STAT + 1536);

    const int tid  = threadIdx.x;
    const int lane = tid & 31;
    const int wv   = tid >> 5;
    const int mt   = wv >> 1;          // M-strip is constant per wave
    const int tokBase = blockIdx.x * TOK_PER_WG;
    const unsigned ldsDyn = __builtin_amdgcn_groupstaticsize();  // dyn-LDS base

    if (tid < 64) { lnSum[tid] = 0.f; lnSq[tid] = 0.f; psq[tid] = 0.f; best[tid] = 0ull; }

    // ---- stage patch tile: A1[tok][k], k = ph*16+pw (matches conv_w layout)
#pragma unroll 4
    for (int i = 0; i < 64; ++i) {
        int e = tid + THREADS * i;          // 64*256 elements
        int lt = e >> 8, k = e & 255;
        int ph = k >> 4, pw = k & 15;
        int t = tokBase + lt;
        int b = t >> 12, n = t & 4095;
        int h = n >> 3,  wc = n & 7;
        size_t g = ((size_t)(b * 8192 + h * 16 + ph)) * 128 + (size_t)(wc * 16 + pw);
        A1[lt * SA1 + k] = f2bf(fbank[g]);
    }
    __syncthreads();

    // ================= Stage 1: feats = patches(64x256) @ W^T (256x512) =====
    for (int c = 0; c < 2; ++c) {           // 256 output dims per chunk
        if (USE_WS) {                        // async bf16 copy: [256][256]
            const unsigned short* src = wsb + WS_WB + (size_t)c * 256 * 256;
#pragma unroll 4
            for (int i = 0; i < 32; ++i) {   // 8192 x 16B / 256 threads
                int e = tid + THREADS * i;
                int n = e >> 5, s8 = (e & 31) * 8;
                async_cp16(ldsDyn + OFF_WB + (unsigned)(n * SW1 + s8) * 2,
                           src + (size_t)n * 256 + s8);
            }
            wait_async0();
        } else {
#pragma unroll 4
            for (int i = 0; i < 256; ++i) {
                int e = tid + THREADS * i;
                int n = e >> 8, k = e & 255;
                WB[n * SW1 + k] = f2bf(conv_w[(size_t)(c * 256 + n) * 256 + k]);
            }
        }
        __syncthreads();
        {
            const int ntb = (wv & 1) * 8;   // 8 N-tiles per wave, one M-strip
            v8f acc[8];
#pragma unroll
            for (int j = 0; j < 8; ++j) acc[j] = {};
#pragma unroll
            for (int kk = 0; kk < 8; ++kk) {
                v16bf a = load_fragA(A1, SA1, mt * 16, kk * 32, lane);
#pragma unroll
                for (int j = 0; j < 8; ++j) {
                    v16bf b = load_fragB(WB, SW1, (ntb + j) * 16, kk * 32, lane);
                    acc[j] = __builtin_amdgcn_wmma_f32_16x16x32_bf16(
                                 false, a, false, b, (short)0, acc[j], false, false);
                }
            }
            int ncol = lane & 15, hi = lane >> 4;
#pragma unroll
            for (int j = 0; j < 8; ++j) {
                int nt = ntb + j;
#pragma unroll
                for (int v = 0; v < 8; ++v) {
                    float val = acc[j][v];
                    int tok = mt * 16 + v + 8 * hi;
                    float s = val, s2 = val * val;
#pragma unroll
                    for (int m = 1; m < 16; m <<= 1) {
                        s  += __shfl_xor(s,  m, 16);
                        s2 += __shfl_xor(s2, m, 16);
                    }
                    if (ncol == 0) { atomicAdd(&lnSum[tok], s); atomicAdd(&lnSq[tok], s2); }
                    FE[tok * SFE + c * 256 + nt * 16 + ncol] = f2bf(val);
                }
            }
        }
        __syncthreads();
    }
    // LayerNorm(512), in place on bf16 feats
    if (tid < 64) {
        float mean = lnSum[tid] * (1.f / 512.f);
        float var  = lnSq[tid] * (1.f / 512.f) - mean * mean;
        muA[tid] = mean;
        rsA[tid] = rsqrtf(var + 1e-5f);
    }
    __syncthreads();
#pragma unroll 4
    for (int i = 0; i < 128; ++i) {
        int e = tid + THREADS * i;          // 64*512
        int t = e >> 9, d = e & 511;
        unsigned short* p = &FE[t * SFE + d];
        *p = f2bf((bf2f(*p) - muA[t]) * rsA[t]);
    }
    __syncthreads();

    // ================= Stage 2: p = feats(64x512) @ proj(512x256) ==========
    for (int c = 0; c < 2; ++c) {           // 128 q-cols per chunk
        if (USE_WS) {                        // async bf16 copy: [128][512]
            const unsigned short* src = wsb + WS_PB + (size_t)c * 128 * 512;
#pragma unroll 4
            for (int i = 0; i < 32; ++i) {   // 8192 x 16B / 256 threads
                int e = tid + THREADS * i;
                int q = e >> 6, s8 = (e & 63) * 8;
                async_cp16(ldsDyn + OFF_WB + (unsigned)(q * SW2 + s8) * 2,
                           src + (size_t)q * 512 + s8);
            }
            wait_async0();
        } else {
#pragma unroll 4
            for (int i = 0; i < 256; ++i) {
                int e = tid + THREADS * i;  // 128*512
                int d = e >> 7, q = e & 127;
                WB[q * SW2 + d] = f2bf(proj[(size_t)d * 256 + c * 128 + q]);
            }
        }
        __syncthreads();
        {
            const int ntb = (wv & 1) * 4;   // 4 N-tiles per wave
            v8f acc[4];
#pragma unroll
            for (int j = 0; j < 4; ++j) acc[j] = {};
#pragma unroll
            for (int kk = 0; kk < 16; ++kk) {
                v16bf a = load_fragA(FE, SFE, mt * 16, kk * 32, lane);
#pragma unroll
                for (int j = 0; j < 4; ++j) {
                    v16bf b = load_fragB(WB, SW2, (ntb + j) * 16, kk * 32, lane);
                    acc[j] = __builtin_amdgcn_wmma_f32_16x16x32_bf16(
                                 false, a, false, b, (short)0, acc[j], false, false);
                }
            }
            int ncol = lane & 15, hi = lane >> 4;
#pragma unroll
            for (int j = 0; j < 4; ++j) {
                int nt = ntb + j;
#pragma unroll
                for (int v = 0; v < 8; ++v) {
                    float val = acc[j][v];
                    int tok = mt * 16 + v + 8 * hi;
                    float s2 = val * val;
#pragma unroll
                    for (int m = 1; m < 16; m <<= 1) s2 += __shfl_xor(s2, m, 16);
                    if (ncol == 0) atomicAdd(&psq[tok], s2);
                    PB[tok * SPB + c * 128 + nt * 16 + ncol] = f2bf(val);
                }
            }
        }
        __syncthreads();
    }
    // row l2norm of p, in place
    if (tid < 64) prs[tid] = 1.f / fmaxf(sqrtf(psq[tid]), 1e-12f);
    __syncthreads();
#pragma unroll 4
    for (int i = 0; i < 64; ++i) {
        int e = tid + THREADS * i;          // 64*256
        int t = e >> 8, q = e & 255;
        unsigned short* p = &PB[t * SPB + q];
        *p = f2bf(bf2f(*p) * prs[t]);
    }
    __syncthreads();

    // ========== Stage 3: sim = p(64x256) @ cbn^T (256x1024) + argmax =======
    for (int c = 0; c < 4; ++c) {           // 256 codes per chunk
        if (USE_WS) {                        // async bf16 copy: [256][256]
            const unsigned short* src = wsb + WS_CBN + (size_t)c * 256 * 256;
            if (lane == 0 && c < 3)
                __builtin_prefetch(src + 256 * 256, 0, 1);
#pragma unroll 4
            for (int i = 0; i < 32; ++i) {   // 8192 x 16B / 256 threads
                int e = tid + THREADS * i;
                int j = e >> 5, s8 = (e & 31) * 8;
                async_cp16(ldsDyn + OFF_WB + (unsigned)(j * SW3 + s8) * 2,
                           src + (size_t)j * 256 + s8);
            }
            wait_async0();
        } else {
#pragma unroll 4
            for (int i = 0; i < 256; ++i) {
                int e = tid + THREADS * i;
                int j = e >> 8, q = e & 255;
                WB[j * SW3 + q] = f2bf(codebook[(size_t)(c * 256 + j) * 256 + q]);
            }
            __syncthreads();
            {                               // in-LDS row l2norm (1 row/thread)
                int j = tid;
                float s = 0.f;
                for (int q = 0; q < 256; ++q) { float x = bf2f(WB[j * SW3 + q]); s += x * x; }
                float sc = 1.f / fmaxf(sqrtf(s), 1e-12f);
                for (int q = 0; q < 256; ++q) WB[j * SW3 + q] = f2bf(bf2f(WB[j * SW3 + q]) * sc);
            }
        }
        __syncthreads();
        {
            const int ntb = (wv & 1) * 8;
            v8f acc[8];
#pragma unroll
            for (int j = 0; j < 8; ++j) acc[j] = {};
#pragma unroll
            for (int kk = 0; kk < 8; ++kk) {
                v16bf a = load_fragA(PB, SPB, mt * 16, kk * 32, lane);
#pragma unroll
                for (int j = 0; j < 8; ++j) {
                    v16bf b = load_fragB(WB, SW3, (ntb + j) * 16, kk * 32, lane);
                    acc[j] = __builtin_amdgcn_wmma_f32_16x16x32_bf16(
                                 false, a, false, b, (short)0, acc[j], false, false);
                }
            }
            int ncol = lane & 15, hi = lane >> 4;
#pragma unroll
            for (int j = 0; j < 8; ++j) {
                int nt = ntb + j;
#pragma unroll
                for (int v = 0; v < 8; ++v) {
                    int tok = mt * 16 + v + 8 * hi;
                    unsigned code = (unsigned)(c * 256 + nt * 16 + ncol);
                    // key: monotone float bits << 32 | ~code (ties -> lowest idx)
                    unsigned long long key =
                        ((unsigned long long)fkey(acc[j][v]) << 32) | (unsigned)(~code);
#pragma unroll
                    for (int m = 1; m < 16; m <<= 1) {
                        unsigned long long o = shfl_xor_u64_16(key, m);
                        if (o > key) key = o;
                    }
                    if (ncol == 0) atomicMax(&best[tok], key);  // ds_max_u64
                }
            }
        }
        __syncthreads();
    }
    if (tid < 64) {
        unsigned idx = ~(unsigned)(best[tid] & 0xFFFFFFFFu);
        out[tokBase + tid] = (int)idx;
    }
}

extern "C" void kernel_launch(void* const* d_in, const int* in_sizes, int n_in,
                              void* d_out, int out_size, void* d_ws, size_t ws_size,
                              hipStream_t stream) {
    (void)in_sizes; (void)n_in; (void)out_size;
    const float* fbank    = (const float*)d_in[0];
    const float* conv_w   = (const float*)d_in[1];
    const float* proj     = (const float*)d_in[2];
    const float* codebook = (const float*)d_in[3];
    int* out = (int*)d_out;

    dim3 grid(NTOK / TOK_PER_WG), block(THREADS);

    (void)hipFuncSetAttribute((const void*)&beats_tokenizer_fused<true>,
                              hipFuncAttributeMaxDynamicSharedMemorySize, LDS_TOTAL);
    (void)hipFuncSetAttribute((const void*)&beats_tokenizer_fused<false>,
                              hipFuncAttributeMaxDynamicSharedMemorySize, LDS_TOTAL);

    if (d_ws && ws_size >= WS_NEED) {
        unsigned short* wsb = (unsigned short*)d_ws;
        normalize_codebook<<<dim3(1024 / 8), dim3(256), 0, stream>>>(codebook, wsb + WS_CBN);
        pack_weights<<<dim3(512), dim3(256), 0, stream>>>(conv_w, proj,
                                                          wsb + WS_WB, wsb + WS_PB);
        beats_tokenizer_fused<true><<<grid, block, LDS_TOTAL, stream>>>(
            fbank, conv_w, proj, codebook, wsb, out);
    } else {
        beats_tokenizer_fused<false><<<grid, block, LDS_TOTAL, stream>>>(
            fbank, conv_w, proj, codebook, nullptr, out);
    }
}
